// RNNModel_72507637891437
// MI455X (gfx1250) — matrix operand order repaired and checked by reading, not compiled, not run on hardware
//
#include <hip/hip_runtime.h>
#include <hip/hip_bf16.h>
#include <math.h>

// Problem constants (match reference)
#define RVOCAB 2048
#define RHID   1024
#define RT     256
#define RB     128

typedef __attribute__((ext_vector_type(16))) _Float16 v16h;
typedef __attribute__((ext_vector_type(8)))  _Float16 v8h;
typedef __attribute__((ext_vector_type(8)))  float    v8f;

__device__ __forceinline__ float fast_tanh(float x) {
#if __has_builtin(__builtin_amdgcn_tanhf)
  return __builtin_amdgcn_tanhf(x);   // V_TANH_F32 (CDNA5 TRANS op)
#else
  return tanhf(x);
#endif
}

// ---------------------------------------------------------------------------
// Fragment loaders for v_wmma_f32_16x16x32_f16 (wave32).
//
// A (16xK=32, f16): lane l (l&15 = row m). lanes 0-15 hold K 0..7 in v0..v3 and
// K 16..23 in v4..v7; lanes 16-31 hold K 8..15 / 24..31. Per lane: two
// contiguous 8-halfword (16B) chunks at k + ksel*8 and k + 16 + ksel*8.
//
// B (K=32 x 16, f16), operand pre-transposed to N-major (BT[n][k]): lane l
// holds column n = l&15, same K order as A: chunks at k + ksel*8 and k + 16 + ksel*8.
// ---------------------------------------------------------------------------
__device__ __forceinline__ v16h load_a_frag(const _Float16* __restrict__ arow,
                                            int k, int ksel) {
  v8h lo = *reinterpret_cast<const v8h*>(arow + k + ksel * 8);
  v8h hi = *reinterpret_cast<const v8h*>(arow + k + 16 + ksel * 8);
  return __builtin_shufflevector(lo, hi, 0, 1, 2, 3, 4, 5, 6, 7,
                                 8, 9, 10, 11, 12, 13, 14, 15);
}

__device__ __forceinline__ v16h load_b_frag(const _Float16* __restrict__ brow,
                                            int k, int ksel) {
  v8h lo = *reinterpret_cast<const v8h*>(brow + k + ksel * 8);
  v8h hi = *reinterpret_cast<const v8h*>(brow + k + 16 + ksel * 8);
  return __builtin_shufflevector(lo, hi, 0, 1, 2, 3, 4, 5, 6, 7,
                                 8, 9, 10, 11, 12, 13, 14, 15);
}

// ---------------------------------------------------------------------------
// Prep: f32 (K x N, row-major) -> f16 transposed (N x K). Coalesced writes.
// ---------------------------------------------------------------------------
__global__ void cvt_transpose_f16(const float* __restrict__ src,
                                  _Float16* __restrict__ dst,
                                  int K, int N) {
  long long idx = (long long)blockIdx.x * blockDim.x + threadIdx.x;
  long long total = (long long)K * N;
  if (idx < total) {
    int n = (int)(idx / K);
    int k = (int)(idx % K);
    dst[idx] = (_Float16)src[(long long)k * N + n];
  }
}

__global__ void cvt_f16(const float* __restrict__ src,
                        _Float16* __restrict__ dst, int n) {
  int idx = blockIdx.x * blockDim.x + threadIdx.x;
  if (idx < n) dst[idx] = (_Float16)src[idx];
}

// ---------------------------------------------------------------------------
// Embedding precompute (off the sequential critical path):
//   E16[(t*B + b)*HID + h] = f16( Wx[tokens[b][t]][h] + bias[h] )
// One thread per element of the (T*B, HID) tensor; coalesced gather rows.
// ---------------------------------------------------------------------------
__global__ void embed_kernel(const int*   __restrict__ tokens, // (B, T)
                             const float* __restrict__ Wx,     // (VOCAB, HID)
                             const float* __restrict__ bias,   // (HID,)
                             _Float16*    __restrict__ E16) {  // (T*B, HID)
  size_t idx = (size_t)blockIdx.x * blockDim.x + threadIdx.x;
  const int h = (int)(idx & (RHID - 1));
  const size_t r = idx >> 10;           // r = t*B + b
  const int t = (int)(r >> 7);          // / RB (128)
  const int b = (int)(r & (RB - 1));
  const int tok = tokens[(size_t)b * RT + t];
  E16[idx] = (_Float16)(Wx[(size_t)tok * RHID + h] + bias[h]);
}

// ---------------------------------------------------------------------------
// One recurrent step:  Hout = tanh( E[t] + Hprev @ Wh )
// Block = 128 threads (4 waves). Each wave computes a 16x32 slab (2 col
// tiles) reusing one A fragment per K-chunk -> 6 loads : 2 WMMA.
// E values are loaded BEFORE the K-loop so they overlap the GEMM; the
// post-WMMA path is pure ALU (tanh + stores).
// Grid: (HID/128, B/16) = (8, 8), 256 waves per step.
// ---------------------------------------------------------------------------
__global__ void rnn_step_kernel(const _Float16* __restrict__ Hprev, // B x HID
                                const _Float16* __restrict__ WhT,   // HID(N) x HID(K)
                                const _Float16* __restrict__ Et,    // B x HID (this step's E)
                                _Float16*       __restrict__ Hout,  // B x HID (f16)
                                float*          __restrict__ hfinal,// B x HID (f32)
                                int write_final) {
  const int wave = threadIdx.x >> 5;
  const int lane = threadIdx.x & 31;
  const int ksel = lane >> 4;
  const int m    = lane & 15;

  const int col0 = blockIdx.x * 128 + wave * 32;
  const int row0 = blockIdx.y * 16;
  const int c0 = col0 + m;
  const int c1 = col0 + 16 + m;

  // Issue the (address-independent) epilogue loads first; they complete
  // while the K-loop runs.
  float e0[8], e1[8];
#pragma unroll
  for (int j = 0; j < 8; ++j) {
    const int r = row0 + j + 8 * ksel;
    e0[j] = (float)Et[(size_t)r * RHID + c0];
    e1[j] = (float)Et[(size_t)r * RHID + c1];
  }

  const _Float16* arow  = Hprev + (size_t)(row0 + m) * RHID;
  const _Float16* brow0 = WhT + (size_t)c0 * RHID;
  const _Float16* brow1 = WhT + (size_t)c1 * RHID;

  v8f acc0 = {};
  v8f acc1 = {};
#pragma unroll 4
  for (int k = 0; k < RHID; k += 32) {
    v16h a  = load_a_frag(arow, k, ksel);
    v16h b0 = load_b_frag(brow0, k, ksel);
    v16h b1 = load_b_frag(brow1, k, ksel);
    acc0 = __builtin_amdgcn_wmma_f32_16x16x32_f16(false, a, false, b0,
                                                  (short)0, acc0, false, false);
    acc1 = __builtin_amdgcn_wmma_f32_16x16x32_f16(false, a, false, b1,
                                                  (short)0, acc1, false, false);
  }

#pragma unroll
  for (int j = 0; j < 8; ++j) {
    const int r = row0 + j + 8 * ksel;   // C/D layout: VGPR j -> row j + 8*(lane>>4)
    float h0 = fast_tanh(acc0[j] + e0[j]);
    float h1 = fast_tanh(acc1[j] + e1[j]);
    Hout[(size_t)r * RHID + c0] = (_Float16)h0;
    Hout[(size_t)r * RHID + c1] = (_Float16)h1;
    if (write_final) {
      hfinal[(size_t)r * RHID + c0] = h0;
      hfinal[(size_t)r * RHID + c1] = h1;
    }
  }
}

// ---------------------------------------------------------------------------
// Output projection:  out = Y(T*B x HID) @ Wd(HID x VOCAB) + bd, f32 out.
// Block = 128 threads (4 waves). Each wave computes a 16x64 slab (4 col
// tiles), reusing one A fragment per K-chunk -> 10 loads : 4 WMMA.
// Output is streamed with non-temporal stores (write-once, 256 MB > L2).
// Grid: (VOCAB/256, T*B/16) = (8, 2048).
// ---------------------------------------------------------------------------
__global__ void proj_kernel(const _Float16* __restrict__ Y,    // (T*B) x HID
                            const _Float16* __restrict__ WdT,  // VOCAB(N) x HID(K)
                            const float*    __restrict__ bd,   // VOCAB
                            float*          __restrict__ out)  // (T*B) x VOCAB
{
  const int wave = threadIdx.x >> 5;
  const int lane = threadIdx.x & 31;
  const int ksel = lane >> 4;
  const int m    = lane & 15;

  const int col0 = blockIdx.x * 256 + wave * 64;
  const int row0 = blockIdx.y * 16;

  const _Float16* arow = Y + (size_t)(row0 + m) * RHID;
  const _Float16* brow[4];
#pragma unroll
  for (int jt = 0; jt < 4; ++jt)
    brow[jt] = WdT + (size_t)(col0 + 16 * jt + m) * RHID;

  v8f acc[4] = {};
#pragma unroll 2
  for (int k = 0; k < RHID; k += 32) {
    v16h a = load_a_frag(arow, k, ksel);
#pragma unroll
    for (int jt = 0; jt < 4; ++jt) {
      v16h b = load_b_frag(brow[jt], k, ksel);
      acc[jt] = __builtin_amdgcn_wmma_f32_16x16x32_f16(false, a, false, b,
                                                       (short)0, acc[jt],
                                                       false, false);
    }
  }

#pragma unroll
  for (int jt = 0; jt < 4; ++jt) {
    const int col = col0 + 16 * jt + m;
    const float bcol = bd[col];
    v8f av = acc[jt];
#pragma unroll
    for (int j = 0; j < 8; ++j) {
      const int r = row0 + j + 8 * ksel;
      __builtin_nontemporal_store(av[j] + bcol, &out[(size_t)r * RVOCAB + col]);
    }
  }
}

// ---------------------------------------------------------------------------
// Host-side launch. Workspace layout (all f16, ~138 MB):
//   WhT  : HID  x HID        (transposed Wh)
//   WdT  : VOCAB x HID       (transposed Wd)
//   S16  : B x HID           (f16 initial state)
//   Y16  : T x B x HID       (f16 hidden states, feeds projection GEMM)
//   E16  : T x B x HID       (f16 precomputed Wx[tok] + b)
// d_out: [ T*B*VOCAB f32 out | B*HID f32 h_final ]
// ---------------------------------------------------------------------------
extern "C" void kernel_launch(void* const* d_in, const int* in_sizes, int n_in,
                              void* d_out, int out_size, void* d_ws, size_t ws_size,
                              hipStream_t stream) {
  const int*   inputs = (const int*)  d_in[0];  // (B, T)
  const float* state  = (const float*)d_in[1];  // (B, HID)
  const float* Wx     = (const float*)d_in[2];  // (VOCAB, HID)
  const float* Wh     = (const float*)d_in[3];  // (HID, HID)
  const float* bias   = (const float*)d_in[4];  // (HID,)
  const float* Wd     = (const float*)d_in[5];  // (HID, VOCAB)
  const float* bd     = (const float*)d_in[6];  // (VOCAB,)

  float* out    = (float*)d_out;
  float* hfinal = out + (size_t)RT * RB * RVOCAB;

  _Float16* WhT = (_Float16*)d_ws;
  _Float16* WdT = WhT + (size_t)RHID * RHID;
  _Float16* S16 = WdT + (size_t)RHID * RVOCAB;
  _Float16* Y16 = S16 + (size_t)RB * RHID;
  _Float16* E16 = Y16 + (size_t)RT * RB * RHID;

  // Weight conversion / transposition (f32 -> f16, N-major for B-fragments)
  {
    long long n1 = (long long)RHID * RHID;
    cvt_transpose_f16<<<(unsigned)((n1 + 255) / 256), 256, 0, stream>>>(Wh, WhT, RHID, RHID);
    long long n2 = (long long)RHID * RVOCAB;
    cvt_transpose_f16<<<(unsigned)((n2 + 255) / 256), 256, 0, stream>>>(Wd, WdT, RHID, RVOCAB);
    int n3 = RB * RHID;
    cvt_f16<<<(n3 + 255) / 256, 256, 0, stream>>>(state, S16, n3);
  }

  // Precompute all step inputs E[t] = Wx[tok] + b (parallel, off the chain).
  {
    size_t ne = (size_t)RT * RB * RHID;
    embed_kernel<<<(unsigned)(ne / 256), 256, 0, stream>>>(inputs, Wx, bias, E16);
  }

  // Sequential recurrence: 256 dependent small GEMMs (graph nodes).
  dim3 stepGrid(RHID / 128, RB / 16);   // (8, 8) blocks of 4 waves
  for (int t = 0; t < RT; ++t) {
    const _Float16* Hp = (t == 0) ? S16 : (Y16 + (size_t)(t - 1) * RB * RHID);
    rnn_step_kernel<<<stepGrid, 128, 0, stream>>>(
        Hp, WhT, E16 + (size_t)t * RB * RHID,
        Y16 + (size_t)t * RB * RHID, hfinal, (t == RT - 1) ? 1 : 0);
  }

  // Output projection: (32768 x 1024) @ (1024 x 2048) + bd
  dim3 projGrid(RVOCAB / 256, (RT * RB) / 16);  // (8, 2048)
  proj_kernel<<<projGrid, 128, 0, stream>>>(Y16, WdT, bd, out);
}
